// ShortTermMemory_34187939676370
// MI455X (gfx1250) — matrix operand order repaired and checked by reading, not compiled, not run on hardware
//
#include <hip/hip_runtime.h>

#define D     2048
#define MEMN  64
#define MTOT  32768
#define K2    4096
#define EPS   1e-5f

typedef __attribute__((ext_vector_type(16))) __bf16 v16bf;
typedef __attribute__((ext_vector_type(8)))  float  v8f;

union Frag16 {
    v16bf v;
    uint4 q[2];
    unsigned short u[16];
};

__device__ __forceinline__ unsigned short f2bf(float f) {
    union { float f; unsigned int u; } in; in.f = f;
    unsigned int u = in.u;
    unsigned int r = u + 0x7FFFu + ((u >> 16) & 1u);
    return (unsigned short)(r >> 16);
}

__device__ __forceinline__ v8f wmma_bf16(const Frag16& a, const Frag16& b, v8f c) {
    return __builtin_amdgcn_wmma_f32_16x16x32_bf16(false, a.v, false, b.v, (short)0, c, false, false);
}

// 16-byte async global->LDS copy (CDNA5 GLOBAL_LOAD_ASYNC_TO_LDS_B128)
__device__ __forceinline__ void async_copy16(unsigned int lds_off, const void* gptr) {
    asm volatile("global_load_async_to_lds_b128 %0, %1, off"
                 :: "v"(lds_off), "v"(gptr) : "memory");
}
__device__ __forceinline__ void wait_async0() {
    asm volatile("s_wait_asynccnt 0" ::: "memory");
}

// ---------------------------------------------------------------------------
// Prep A: Abig[row][0..2047] = bf16(x[row][0..2047])   (8 elems / thread)
// ---------------------------------------------------------------------------
__global__ __launch_bounds__(256) void cvt_x_kernel(
    const float* __restrict__ x, unsigned short* __restrict__ Abig) {
    long long i = ((long long)blockIdx.x * 256 + threadIdx.x) * 8;
    long long row = i >> 11;
    long long d   = i & 2047;
    const float4* p = (const float4*)(x + i);
    float4 f0 = p[0], f1 = p[1];
    unsigned p0 = f2bf(f0.x) | ((unsigned)f2bf(f0.y) << 16);
    unsigned p1 = f2bf(f0.z) | ((unsigned)f2bf(f0.w) << 16);
    unsigned p2 = f2bf(f1.x) | ((unsigned)f2bf(f1.y) << 16);
    unsigned p3 = f2bf(f1.z) | ((unsigned)f2bf(f1.w) << 16);
    uint4 q = {p0, p1, p2, p3};
    *(uint4*)(Abig + row * K2 + d) = q;
}

// ---------------------------------------------------------------------------
// Prep W: dst[n*K + k] = bf16(src[k*N + n])   (transpose + f32->bf16)
// ---------------------------------------------------------------------------
__global__ __launch_bounds__(256) void transpose_cvt_kernel(
    const float* __restrict__ src, unsigned short* __restrict__ dst, int K, int N) {
    long long idx = (long long)blockIdx.x * 256 + threadIdx.x;
    if (idx >= (long long)K * N) return;
    int n = (int)(idx / K);
    int k = (int)(idx % K);
    dst[idx] = f2bf(src[(long long)k * N + n]);
}

// ---------------------------------------------------------------------------
// Stage 1: gates = sigmoid(xb*WgT + bg) [16 x 64], r = gates*mem -> Abig[:,2048:]
// One block = 16 tokens, 128 threads = 4 waves.
// ---------------------------------------------------------------------------
__global__ __launch_bounds__(128) void gates_retrieve_kernel(
    unsigned short* __restrict__ Abig,        // [32768][4096] bf16; reads cols 0..2047, writes 2048..4095
    const float* __restrict__ bg,
    const unsigned short* __restrict__ WgT,   // [64][2048] bf16 (N-major)
    const unsigned short* __restrict__ memT) {// [2048][64] bf16 (N-major)
    const int m0   = blockIdx.x * 16;
    const int t    = threadIdx.x;
    const int w    = t >> 5;
    const int lane = t & 31;
    const int l    = lane & 15;
    const int hi   = lane >> 4;

    __shared__ unsigned short gtile[16][64];

    // ---- gates GEMM: M=16, N=16 (this wave), K=2048
    v8f acc = {0.f,0.f,0.f,0.f,0.f,0.f,0.f,0.f};
    const int n = w * 16 + l;
    const unsigned short* rowp = Abig + (long long)(m0 + l) * K2;
    for (int k0 = 0; k0 < D; k0 += 32) {
        Frag16 a;
        a.q[0] = *(const uint4*)(rowp + k0 + hi * 16);
        a.q[1] = *(const uint4*)(rowp + k0 + hi * 16 + 8);
        Frag16 b;
        b.v = *(const v16bf*)(WgT + (long long)n * D + k0 + hi * 16);
        acc = wmma_bf16(a, b, acc);
    }
    const float bias = bg[n];
#pragma unroll
    for (int i = 0; i < 8; ++i) {
        float g = __builtin_amdgcn_rcpf(1.0f + __expf(-(acc[i] + bias)));
        gtile[hi * 8 + i][n] = f2bf(g);
    }
    __syncthreads();

    // ---- re-layout gates as A fragments (K = 64 -> two k-steps)
    Frag16 ar[2];
#pragma unroll
    for (int s = 0; s < 2; ++s) {
        ar[s].q[0] = *(const uint4*)&gtile[l][s * 32 + hi * 16];
        ar[s].q[1] = *(const uint4*)&gtile[l][s * 32 + hi * 16 + 8];
    }

    // ---- retrieval GEMM: this wave covers 512 columns = 32 n-tiles
    unsigned short* r_out = Abig + D;  // columns 2048..4095
#pragma unroll 4
    for (int nt = 0; nt < 32; ++nt) {
        const int nb = w * 512 + nt * 16;
        v8f racc = {0.f,0.f,0.f,0.f,0.f,0.f,0.f,0.f};
#pragma unroll
        for (int s = 0; s < 2; ++s) {
            Frag16 b;
            b.v = *(const v16bf*)(memT + (long long)(nb + l) * MEMN + s * 32 + hi * 16);
            racc = wmma_bf16(ar[s], b, racc);
        }
        const int col = nb + l;
#pragma unroll
        for (int i = 0; i < 8; ++i)
            r_out[(long long)(m0 + hi * 8 + i) * K2 + col] = f2bf(racc[i]);
    }
}

// ---------------------------------------------------------------------------
// Stage 2: enhanced = Abig * WuT + bu     C[32768, 2048], K = 4096
// Block tile 128x128, BK=64, 256 threads = 8 waves (4M x 2N).
// Both A and B tiles staged via async global->LDS (full 128B line per row),
// double-buffered, one tile-pair in flight. Inner loop: LDS + WMMA only.
// blockIdx.x = N (16), blockIdx.y = M (256) so A stripes + WuT stay L2-hot.
// ---------------------------------------------------------------------------
#define BM  128
#define BN  128
#define BKK 64

__global__ __launch_bounds__(256) void gemm_bias_kernel(
    const unsigned short* __restrict__ Abig,  // [32768][4096] bf16
    const unsigned short* __restrict__ WuT,   // [2048][4096] bf16 (N-major)
    const float* __restrict__ bu,
    float* __restrict__ out) {
    __shared__ unsigned short As[2][BM][BKK];   // 32 KB
    __shared__ unsigned short Bs[2][BN][BKK];   // 32 KB (column-major over N)

    const int t    = threadIdx.x;
    const int n0   = blockIdx.x * BN;
    const int m0   = blockIdx.y * BM;
    const int w    = t >> 5;
    const int lane = t & 31;
    const int l    = lane & 15;
    const int hi   = lane >> 4;
    const int wm   = w & 3;
    const int wn   = w >> 2;

    const int crow = t >> 1;   // 0..127 (row of A / column of B to copy)
    const int chalf = t & 1;   // which 64B half of the 128B line

    v8f acc[2][4];
#pragma unroll
    for (int i = 0; i < 2; ++i)
#pragma unroll
        for (int j = 0; j < 4; ++j)
            acc[i][j] = (v8f){0.f,0.f,0.f,0.f,0.f,0.f,0.f,0.f};

    auto issueTiles = [&](int buf, int k0) {
        {   // A: 64 bytes per thread
            const unsigned short* g = Abig + (long long)(m0 + crow) * K2 + k0 + chalf * 32;
            unsigned lds0 = (unsigned)(unsigned long long)&As[buf][crow][chalf * 32];
            async_copy16(lds0,      g);
            async_copy16(lds0 + 16, g + 8);
            async_copy16(lds0 + 32, g + 16);
            async_copy16(lds0 + 48, g + 24);
        }
        {   // B: 64 bytes per thread
            const unsigned short* g = WuT + (long long)(n0 + crow) * K2 + k0 + chalf * 32;
            unsigned lds0 = (unsigned)(unsigned long long)&Bs[buf][crow][chalf * 32];
            async_copy16(lds0,      g);
            async_copy16(lds0 + 16, g + 8);
            async_copy16(lds0 + 32, g + 16);
            async_copy16(lds0 + 48, g + 24);
        }
    };

    // preload tile 0
    issueTiles(0, 0);
    wait_async0();
    __syncthreads();

    const int NK = K2 / BKK;  // 64
    for (int kt = 0; kt < NK; ++kt) {
        const int buf = kt & 1;
        const bool have_next = (kt + 1 < NK);
        if (have_next) issueTiles(buf ^ 1, (kt + 1) * BKK);

#pragma unroll
        for (int s = 0; s < 2; ++s) {
            // A fragments from LDS
            Frag16 a[2];
#pragma unroll
            for (int i = 0; i < 2; ++i) {
                const int row = wm * 32 + i * 16 + l;
                a[i].q[0] = *(const uint4*)&As[buf][row][s * 32 + hi * 16];
                a[i].q[1] = *(const uint4*)&As[buf][row][s * 32 + hi * 16 + 8];
            }
            // B fragments from LDS + 8 WMMAs
#pragma unroll
            for (int j = 0; j < 4; ++j) {
                const int col = wn * 64 + j * 16 + l;
                Frag16 b;
                b.q[0] = *(const uint4*)&Bs[buf][col][s * 32 + hi * 16];
                b.q[1] = *(const uint4*)&Bs[buf][col][s * 32 + hi * 16 + 8];
#pragma unroll
                for (int i = 0; i < 2; ++i)
                    acc[i][j] = wmma_bf16(a[i], b, acc[i][j]);
            }
        }

        if (have_next) {
            wait_async0();
            __syncthreads();
        }
    }

    // epilogue: + bias, store f32
#pragma unroll
    for (int j = 0; j < 4; ++j) {
        const int col = n0 + wn * 64 + j * 16 + l;
        const float bias = bu[col];
#pragma unroll
        for (int i = 0; i < 2; ++i) {
            const int rbase = m0 + wm * 32 + i * 16 + hi * 8;
#pragma unroll
            for (int q = 0; q < 8; ++q)
                out[(long long)(rbase + q) * D + col] = acc[i][j][q] + bias;
        }
    }
}

// ---------------------------------------------------------------------------
// Stage 3: row LayerNorm in place on d_out. One block per row (2048 f32).
// ---------------------------------------------------------------------------
__global__ __launch_bounds__(256) void layernorm_kernel(
    float* __restrict__ out, const float* __restrict__ gamma,
    const float* __restrict__ beta) {
    const int row = blockIdx.x;
    float* h = out + (long long)row * D;
    const int t = threadIdx.x;

    float4 v0 = ((const float4*)h)[t * 2 + 0];
    float4 v1 = ((const float4*)h)[t * 2 + 1];
    float s  = v0.x + v0.y + v0.z + v0.w + v1.x + v1.y + v1.z + v1.w;
    float s2 = v0.x*v0.x + v0.y*v0.y + v0.z*v0.z + v0.w*v0.w
             + v1.x*v1.x + v1.y*v1.y + v1.z*v1.z + v1.w*v1.w;

    __shared__ float rs[256], rq[256];
    rs[t] = s; rq[t] = s2;
    __syncthreads();
#pragma unroll
    for (int o = 128; o > 0; o >>= 1) {
        if (t < o) { rs[t] += rs[t + o]; rq[t] += rq[t + o]; }
        __syncthreads();
    }
    const float mu   = rs[0] * (1.f / D);
    const float var  = rq[0] * (1.f / D) - mu * mu;
    const float rstd = rsqrtf(var + EPS);

    const int base = t * 8;
    float4 g0 = ((const float4*)(gamma + base))[0];
    float4 g1 = ((const float4*)(gamma + base))[1];
    float4 b0 = ((const float4*)(beta + base))[0];
    float4 b1 = ((const float4*)(beta + base))[1];
    float4 o0, o1;
    o0.x = (v0.x - mu) * rstd * g0.x + b0.x;
    o0.y = (v0.y - mu) * rstd * g0.y + b0.y;
    o0.z = (v0.z - mu) * rstd * g0.z + b0.z;
    o0.w = (v0.w - mu) * rstd * g0.w + b0.w;
    o1.x = (v1.x - mu) * rstd * g1.x + b1.x;
    o1.y = (v1.y - mu) * rstd * g1.y + b1.y;
    o1.z = (v1.z - mu) * rstd * g1.z + b1.z;
    o1.w = (v1.w - mu) * rstd * g1.w + b1.w;
    ((float4*)h)[t * 2 + 0] = o0;
    ((float4*)h)[t * 2 + 1] = o1;
}

// ---------------------------------------------------------------------------
extern "C" void kernel_launch(void* const* d_in, const int* in_sizes, int n_in,
                              void* d_out, int out_size, void* d_ws, size_t ws_size,
                              hipStream_t stream) {
    const float* x     = (const float*)d_in[0];
    const float* mem   = (const float*)d_in[1];
    const float* Wg    = (const float*)d_in[2];
    const float* bg    = (const float*)d_in[3];
    const float* Wu    = (const float*)d_in[4];
    const float* bu    = (const float*)d_in[5];
    const float* gamma = (const float*)d_in[6];
    const float* beta  = (const float*)d_in[7];
    float* out = (float*)d_out;

    char* ws = (char*)d_ws;
    unsigned short* Abig = (unsigned short*)(ws);                        // [32768][4096] bf16 = 256 MB
    unsigned short* WuT  = (unsigned short*)(ws + 268435456);            // 2048*4096 bf16 = 16 MB
    unsigned short* WgT  = (unsigned short*)(ws + 268435456 + 16777216);           // 64*2048
    unsigned short* memT = (unsigned short*)(ws + 268435456 + 16777216 + 262144);  // 2048*64

    // prep
    cvt_x_kernel<<<(MTOT * D / 8) / 256, 256, 0, stream>>>(x, Abig);
    transpose_cvt_kernel<<<(K2 * D) / 256, 256, 0, stream>>>(Wu, WuT, K2, D);
    transpose_cvt_kernel<<<(D * MEMN) / 256, 256, 0, stream>>>(Wg, WgT, D, MEMN);
    transpose_cvt_kernel<<<(MEMN * D) / 256, 256, 0, stream>>>(mem, memT, MEMN, D);

    // stage 1: gates + retrieval (fills Abig columns 2048..4095)
    gates_retrieve_kernel<<<MTOT / 16, 128, 0, stream>>>(Abig, bg, WgT, memT);

    // stage 2: big GEMM + bias
    dim3 grid(D / BN, MTOT / BM);
    gemm_bias_kernel<<<grid, 256, 0, stream>>>(Abig, WuT, bu, out);

    // stage 3: layernorm in place
    layernorm_kernel<<<MTOT, 256, 0, stream>>>(out, gamma, beta);
}
